// TopkSelfAttention_71090298683453
// MI455X (gfx1250) — matrix operand order repaired and checked by reading, not compiled, not run on hardware
//
#include <hip/hip_runtime.h>
#include <hip/hip_bf16.h>

typedef __attribute__((ext_vector_type(16))) __bf16 v16bf;
typedef __attribute__((ext_vector_type(8)))  __bf16 v8bf;
typedef __attribute__((ext_vector_type(8)))  float  v8f;
typedef __attribute__((ext_vector_type(4)))  unsigned int v4u;
typedef __attribute__((ext_vector_type(8)))  int v8i;
typedef __attribute__((ext_vector_type(4)))  int v4i;

#define HD    64
#define KSEL  512
#define TT    2048      // tokens per (b,h) = KSEL * 4
#define NBH   64        // B * nh
#define HW    4096      // H*W
#define WWID  64        // image width
#define LDSTR 72        // padded LDS row stride (bf16 elements): 64 + 4 DWORDs pad

static __device__ __forceinline__ v16bf combine8(v8bf lo, v8bf hi) {
  v16bf a;
#pragma unroll
  for (int e = 0; e < 8; ++e) { a[e] = lo[e]; a[e + 8] = hi[e]; }
  return a;
}

// Tensor Data Mover: 2D tile (tile1 rows x tile0 elems, 2-byte data) from
// global to LDS, inserting 4 DWORDs of pad every 32 DWORDs (=> row stride 72
// bf16 when tile0 == 64).  Descriptor per CDNA5 ISA ch.8.
static __device__ __forceinline__ void tdm_load_2d(
    const void* gaddr, unsigned lds_off,
    unsigned dim0, unsigned dim1, unsigned tile0, unsigned tile1,
    unsigned long long stride0 /* elements */)
{
  unsigned long long ga = (unsigned long long)(uintptr_t)gaddr;
  v4u g0;
  g0[0] = 1u;                                         // count=1, user D#
  g0[1] = lds_off;                                    // lds_addr (bytes)
  g0[2] = (unsigned)ga;                               // global_addr[31:0]
  g0[3] = (unsigned)((ga >> 32) & 0x1FFFFFFull)       // global_addr[56:32]
        | (2u << 30);                                 // type=2 ("image")
  v8i g1;
  g1[0] = (1 << 16)            // data_size = 1 -> 2 bytes
        | (1 << 20)            // pad_enable
        | (4 << 22)            // pad_interval: 8<<4 = 128 bytes (32 DWORDs)
        | (3 << 25);           // pad_amount: 4 DWORDs (16 bytes)
  g1[1] = (int)((dim0 & 0xFFFFu) << 16);              // tensor_dim0[15:0]
  g1[2] = (int)((dim0 >> 16) | ((dim1 & 0xFFFFu) << 16));
  g1[3] = (int)((dim1 >> 16) | (tile0 << 16));        // tile_dim0
  g1[4] = (int)(tile1 & 0xFFFFu);                     // tile_dim1 (tile_dim2=0)
  g1[5] = (int)(stride0 & 0xFFFFFFFFull);             // tensor_dim0_stride lo
  g1[6] = (int)((stride0 >> 32) & 0xFFFFull);         // stride hi (dim1_stride=0)
  g1[7] = 0;
  v4i z4 = {0, 0, 0, 0};
  v8i z8 = {0, 0, 0, 0, 0, 0, 0, 0};
  __builtin_amdgcn_tensor_load_to_lds(g0, g1, z4, z4, z8, 0);
}

// ---------------------------------------------------------------- zero fill
__global__ void zero_out_kernel(float4* __restrict__ out, int n4) {
  int i = blockIdx.x * blockDim.x + threadIdx.x;
  if (i < n4) out[i] = make_float4(0.f, 0.f, 0.f, 0.f);
}

// ------------------------------------------------ gather + QKV projection
__global__ void __launch_bounds__(128)
qkv_gather_kernel(const float* __restrict__ x, const int* __restrict__ topk,
                  const float* __restrict__ wqkv, const float* __restrict__ bqkv,
                  __bf16* __restrict__ Q, __bf16* __restrict__ K,
                  __bf16* __restrict__ Vt)
{
  __shared__ __bf16 tok[64 * LDSTR];
  __shared__ int ptoff[64];

  const int bh   = blockIdx.y;
  const int b    = bh >> 3, h = bh & 7;
  const int t0   = blockIdx.x * 64;
  const int tid  = threadIdx.x;
  const int lane = tid & 31;
  const int wv   = tid >> 5;
  const int nhalf = (lane >> 4) & 1;
  const int ncol  = lane & 15;

  if (tid < 64) {
    int t  = t0 + tid;
    int p  = topk[bh * KSEL + (t >> 2)];
    int dy = (t >> 1) & 1, dx = t & 1;
    ptoff[tid] = ((p >> 5) * 2 + dy) * WWID + (p & 31) * 2 + dx;
  }
  __syncthreads();

  const float* xb = x + (size_t)(b * 512 + h * 64) * HW;
#pragma unroll
  for (int i = 0; i < 32; ++i) {
    int idx = tid + i * 128;
    int tk  = idx & 63;
    int d   = idx >> 6;
    tok[tk * LDSTR + d] = (__bf16)xb[(size_t)d * HW + ptoff[tk]];
  }
  __syncthreads();

  v16bf bfrag[3][2];
  float bias[3];
#pragma unroll
  for (int jt = 0; jt < 3; ++jt) {
    int jj = (wv * 3 + jt) * 16 + ncol;
    bias[jt] = bqkv[jj];
#pragma unroll
    for (int kk = 0; kk < 2; ++kk) {
      int d0 = kk * 32 + nhalf * 16;
      const float* wp = wqkv + jj * HD + d0;
      v16bf f;
#pragma unroll
      for (int e = 0; e < 16; ++e) f[e] = (__bf16)wp[e];
      bfrag[jt][kk] = f;
    }
  }

#pragma unroll
  for (int mt = 0; mt < 4; ++mt) {
    v16bf afrag[2];
#pragma unroll
    for (int kk = 0; kk < 2; ++kk) {
      int row = mt * 16 + ncol;
      int c0  = kk * 32 + nhalf * 16;
      v8bf lo = *(const v8bf*)&tok[row * LDSTR + c0];
      v8bf hi = *(const v8bf*)&tok[row * LDSTR + c0 + 8];
      afrag[kk] = combine8(lo, hi);
    }
#pragma unroll
    for (int jt = 0; jt < 3; ++jt) {
      v8f acc;
#pragma unroll
      for (int e = 0; e < 8; ++e) acc[e] = bias[jt];
      acc = __builtin_amdgcn_wmma_f32_16x16x32_bf16(false, afrag[0], false,
              bfrag[jt][0], (short)0, acc, false, false);
      acc = __builtin_amdgcn_wmma_f32_16x16x32_bf16(false, afrag[1], false,
              bfrag[jt][1], (short)0, acc, false, false);

      const int jstile = wv * 3 + jt;           // 0..11: uniform per wave
      const int jj     = jstile * 16 + ncol;
      const int tbase  = t0 + mt * 16 + nhalf * 8;
      if (jstile < 4) {                         // ---- Q (pre-scaled)
        __bf16* qp = Q + (size_t)bh * TT * HD + jj;
#pragma unroll
        for (int r = 0; r < 8; ++r)
          qp[(size_t)(tbase + r) * HD] = (__bf16)(acc[r] * 0.125f);
      } else if (jstile < 8) {                  // ---- K row-major
        __bf16* kp = K + (size_t)bh * TT * HD + (jj - 64);
#pragma unroll
        for (int r = 0; r < 8; ++r)
          kp[(size_t)(tbase + r) * HD] = (__bf16)acc[r];
      } else {                                  // ---- V transposed: Vt[d][t]
        __bf16* vp = Vt + (size_t)bh * HD * TT + (size_t)(jj - 128) * TT + tbase;
#pragma unroll
        for (int r = 0; r < 8; ++r)
          vp[r] = (__bf16)acc[r];
      }
    }
  }
}

// --------------------------------------------------------- flash attention
// 4 waves / block, 128 query rows / block.  K/V tiles (64x64 bf16) are
// double-buffered into LDS by the Tensor Data Mover (wave 0 issues, whole
// block consumes), so the DMA for tile st+1 overlaps WMMA work on tile st.
__global__ void __launch_bounds__(128)
flash_attn_kernel(const int* __restrict__ topk,
                  const __bf16* __restrict__ Q, const __bf16* __restrict__ K,
                  const __bf16* __restrict__ Vt, float* __restrict__ out)
{
  __shared__ __bf16 ldsK[2][64 * LDSTR];
  __shared__ __bf16 ldsV[2][64 * LDSTR];
  __shared__ __bf16 pbuf[4][32 * LDSTR];
  __shared__ int ptoff[128];

  const int bh   = blockIdx.y;
  const int b    = bh >> 3, h = bh & 7;
  const int t0   = blockIdx.x * 128;
  const int tid  = threadIdx.x;
  const int lane = tid & 31;
  const int wv   = tid >> 5;
  const int nhalf = (lane >> 4) & 1;
  const int ncol  = lane & 15;

  {
    int t  = t0 + tid;
    int p  = topk[bh * KSEL + (t >> 2)];
    int dy = (t >> 1) & 1, dx = t & 1;
    ptoff[tid] = ((p >> 5) * 2 + dy) * WWID + (p & 31) * 2 + dx;
  }

  const size_t qkbase = (size_t)bh * TT * HD;
  const size_t vtbase = (size_t)bh * HD * TT;
  const int qrow0 = wv * 32;

  // kick off TDM for the first K/V tile
  if (wv == 0) {
    tdm_load_2d(K + qkbase, (unsigned)(uintptr_t)&ldsK[0][0],
                HD, TT, 64, 64, HD);
    tdm_load_2d(Vt + vtbase, (unsigned)(uintptr_t)&ldsV[0][0],
                TT, HD, 64, 64, TT);
  }

  // persistent Q fragments
  v16bf qfrag[2][2];
#pragma unroll
  for (int mt = 0; mt < 2; ++mt) {
    int trow = t0 + qrow0 + mt * 16 + ncol;
#pragma unroll
    for (int kk = 0; kk < 2; ++kk) {
      int c0 = kk * 32 + nhalf * 16;
      v8bf lo = *(const v8bf*)(Q + qkbase + (size_t)trow * HD + c0);
      v8bf hi = *(const v8bf*)(Q + qkbase + (size_t)trow * HD + c0 + 8);
      qfrag[mt][kk] = combine8(lo, hi);
    }
  }

  v8f acc[2][4];
  float mi[2][8], li[2][8];
#pragma unroll
  for (int mt = 0; mt < 2; ++mt) {
#pragma unroll
    for (int dt = 0; dt < 4; ++dt)
#pragma unroll
      for (int e = 0; e < 8; ++e) acc[mt][dt][e] = 0.f;
#pragma unroll
    for (int r = 0; r < 8; ++r) { mi[mt][r] = -1.0e30f; li[mt][r] = 0.f; }
  }

  __bf16* pl = &pbuf[wv][0];

  for (int st = 0; st < 32; ++st) {
    const int buf = st & 1;
    __syncthreads();                      // all waves done with buf^1
    if (wv == 0) {
      if (st + 1 < 32) {                  // prefetch next tile into buf^1
        int snext = (st + 1) * 64;
        tdm_load_2d(K + qkbase + (size_t)snext * HD,
                    (unsigned)(uintptr_t)&ldsK[buf ^ 1][0],
                    HD, TT, 64, 64, HD);
        tdm_load_2d(Vt + vtbase + snext,
                    (unsigned)(uintptr_t)&ldsV[buf ^ 1][0],
                    TT, HD, 64, 64, TT);
        __builtin_amdgcn_s_wait_tensorcnt(2);   // pair for st complete
      } else {
        __builtin_amdgcn_s_wait_tensorcnt(0);
      }
    }
    __syncthreads();                      // publish buf to all waves

    // ---- S = Q @ K^T
    v8f S[2][4];
#pragma unroll
    for (int mt = 0; mt < 2; ++mt)
#pragma unroll
      for (int nt = 0; nt < 4; ++nt)
#pragma unroll
        for (int e = 0; e < 8; ++e) S[mt][nt][e] = 0.f;

#pragma unroll
    for (int nt = 0; nt < 4; ++nt) {
      v16bf kfrag[2];
#pragma unroll
      for (int kk = 0; kk < 2; ++kk) {
        int srow = nt * 16 + ncol;
        int d0   = kk * 32 + nhalf * 16;
        kfrag[kk] = *(const v16bf*)&ldsK[buf][srow * LDSTR + d0];
      }
#pragma unroll
      for (int mt = 0; mt < 2; ++mt) {
        S[mt][nt] = __builtin_amdgcn_wmma_f32_16x16x32_bf16(false, qfrag[mt][0],
                      false, kfrag[0], (short)0, S[mt][nt], false, false);
        S[mt][nt] = __builtin_amdgcn_wmma_f32_16x16x32_bf16(false, qfrag[mt][1],
                      false, kfrag[1], (short)0, S[mt][nt], false, false);
      }
    }

    // ---- online softmax
#pragma unroll
    for (int mt = 0; mt < 2; ++mt) {
      float alpha[8];
#pragma unroll
      for (int r = 0; r < 8; ++r) {
        float rm = fmaxf(fmaxf(S[mt][0][r], S[mt][1][r]),
                         fmaxf(S[mt][2][r], S[mt][3][r]));
        rm = fmaxf(rm, __shfl_xor(rm, 1, 32));
        rm = fmaxf(rm, __shfl_xor(rm, 2, 32));
        rm = fmaxf(rm, __shfl_xor(rm, 4, 32));
        rm = fmaxf(rm, __shfl_xor(rm, 8, 32));
        float mnew = fmaxf(mi[mt][r], rm);
        alpha[r]  = __expf(mi[mt][r] - mnew);
        mi[mt][r] = mnew;
        float rs = 0.f;
#pragma unroll
        for (int nt = 0; nt < 4; ++nt) {
          float pv = __expf(S[mt][nt][r] - mnew);
          S[mt][nt][r] = pv;
          rs += pv;
        }
        rs += __shfl_xor(rs, 1, 32);
        rs += __shfl_xor(rs, 2, 32);
        rs += __shfl_xor(rs, 4, 32);
        rs += __shfl_xor(rs, 8, 32);
        li[mt][r] = li[mt][r] * alpha[r] + rs;
      }
#pragma unroll
      for (int dt = 0; dt < 4; ++dt)
#pragma unroll
        for (int r = 0; r < 8; ++r) acc[mt][dt][r] *= alpha[r];

#pragma unroll
      for (int nt = 0; nt < 4; ++nt)
#pragma unroll
        for (int r = 0; r < 8; ++r) {
          int row = mt * 16 + nhalf * 8 + r;
          pl[row * LDSTR + nt * 16 + ncol] = (__bf16)S[mt][nt][r];
        }
    }
    asm volatile("s_wait_dscnt 0" ::: "memory");   // same-wave LDS RAW

    v16bf pfrag[2][2];
#pragma unroll
    for (int mt = 0; mt < 2; ++mt)
#pragma unroll
      for (int kk = 0; kk < 2; ++kk) {
        int row = mt * 16 + ncol;
        int c0  = kk * 32 + nhalf * 16;
        v8bf lo = *(const v8bf*)&pl[row * LDSTR + c0];
        v8bf hi = *(const v8bf*)&pl[row * LDSTR + c0 + 8];
        pfrag[mt][kk] = combine8(lo, hi);
      }

    // ---- acc += P @ V
#pragma unroll
    for (int dt = 0; dt < 4; ++dt) {
      v16bf vfrag[2];
#pragma unroll
      for (int kk = 0; kk < 2; ++kk) {
        int drow = dt * 16 + ncol;
        int s0   = kk * 32 + nhalf * 16;
        vfrag[kk] = *(const v16bf*)&ldsV[buf][drow * LDSTR + s0];
      }
#pragma unroll
      for (int mt = 0; mt < 2; ++mt) {
        acc[mt][dt] = __builtin_amdgcn_wmma_f32_16x16x32_bf16(false, pfrag[mt][0],
                        false, vfrag[0], (short)0, acc[mt][dt], false, false);
        acc[mt][dt] = __builtin_amdgcn_wmma_f32_16x16x32_bf16(false, pfrag[mt][1],
                        false, vfrag[1], (short)0, acc[mt][dt], false, false);
      }
    }
  }

  // ---- normalize and scatter
  const size_t outb = (size_t)(b * 512 + h * 64) * HW;
#pragma unroll
  for (int mt = 0; mt < 2; ++mt)
#pragma unroll
    for (int r = 0; r < 8; ++r) {
      float inv = 1.0f / li[mt][r];
      int row = qrow0 + mt * 16 + nhalf * 8 + r;
      int po  = ptoff[row];
#pragma unroll
      for (int dt = 0; dt < 4; ++dt) {
        int d = dt * 16 + ncol;
        out[outb + (size_t)d * HW + po] = acc[mt][dt][r] * inv;
      }
    }
}

// ----------------------------------------------------------------- launch
extern "C" void kernel_launch(void* const* d_in, const int* in_sizes, int n_in,
                              void* d_out, int out_size, void* d_ws, size_t ws_size,
                              hipStream_t stream) {
  const float* x    = (const float*)d_in[0];
  const int*   topk = (const int*)d_in[1];
  const float* wqkv = (const float*)d_in[2];
  const float* bqkv = (const float*)d_in[3];

  __bf16* Q  = (__bf16*)d_ws;
  __bf16* K  = Q + (size_t)NBH * TT * HD;
  __bf16* Vt = K + (size_t)NBH * TT * HD;

  int n4 = out_size / 4;
  zero_out_kernel<<<dim3((n4 + 255) / 256), dim3(256), 0, stream>>>(
      (float4*)d_out, n4);
  qkv_gather_kernel<<<dim3(32, 64), dim3(128), 0, stream>>>(
      x, topk, wqkv, bqkv, Q, K, Vt);
  flash_attn_kernel<<<dim3(16, 64), dim3(128), 0, stream>>>(
      topk, Q, K, Vt, (float*)d_out);
}
